// ComplexSpectralAttention_75393855914098
// MI455X (gfx1250) — compile-verified, matrix operand order checked by reading, not run on hardware
//
#include <hip/hip_runtime.h>

// ---------------------------------------------------------------------------
// ComplexSpectralAttention == standard causal MHA, head_dim=128, scale=1/8,
// plus QKV projection GEMMs and an output projection GEMM.
// Shapes: B=2, N=2048, C=1024, H=16, D=128 (per head), tokens NT=4096.
// Strategy: f16 WMMA (v_wmma_f32_16x16x32_f16) everywhere, f32 accumulate.
// Attention streams K/V tiles into LDS with global_load_async_to_lds_b128
// (ASYNCcnt), double-buffered, shared by all 8 waves of the block.
// Workspace (f16 elements), total ~88 MB: xb, WqT/WkT/WvT/WoT, Q, K, Vt, A2.
// ---------------------------------------------------------------------------

typedef __attribute__((ext_vector_type(16))) _Float16     v16h;
typedef __attribute__((ext_vector_type(8)))  float        v8f;
typedef __attribute__((ext_vector_type(4)))  unsigned int v4u;

union Frag {            // one WMMA A/B operand: 16 f16 = 8 VGPRs = 2 x b128
    v16h v;
    v4u  q[2];          // q[0] = K {kb..kb+7}, q[1] = K {kb+16..kb+23}
};

__device__ __forceinline__ unsigned short f2h(float f) {
    union { _Float16 h; unsigned short u; } cv;
    cv.h = (_Float16)f;
    return cv.u;
}

__device__ __forceinline__ v4u ldg128(const unsigned short* p) {
    return *(const v4u*)p;
}

__device__ __forceinline__ v8f wmma16(const Frag& a, const Frag& b, v8f c) {
    return __builtin_amdgcn_wmma_f32_16x16x32_f16(
        false, a.v, false, b.v, (short)0, c, false, false);
}

// Async copy 16B/lane from global to LDS (gfx1250 VGLOBAL, tracked by ASYNCcnt)
__device__ __forceinline__ void async_ld128(unsigned lds_off,
                                            const unsigned short* gptr) {
    unsigned long long ga = (unsigned long long)(size_t)gptr;
    asm volatile("global_load_async_to_lds_b128 %0, %1, off"
                 :: "v"(lds_off), "v"(ga) : "memory");
}

#define SEQ   2048
#define NTOK  4096
#define CIN   1024
#define C2    2048
#define HEADS 16
#define KPAD  136   // 128 + 8 f16 pad (16B) to break LDS bank conflicts
#define VPAD  40    // 32 + 8 f16 pad

// ---------------------------------------------------------------------------
// Kernel 1a: x f32 -> f16
__global__ __launch_bounds__(256) void csa_cvt_kernel(
    const float* __restrict__ x, unsigned short* __restrict__ xb, int n) {
    int i = blockIdx.x * 256 + threadIdx.x;
    if (i < n) xb[i] = f2h(x[i]);
}

// Kernel 1b: W (Krows x Ncols, f32) -> WT (Ncols x Krows, f16)
__global__ __launch_bounds__(256) void csa_cvt_tr_kernel(
    const float* __restrict__ W, unsigned short* __restrict__ WT,
    int Krows, int Ncols) {
    int i = blockIdx.x * 256 + threadIdx.x;
    if (i < Krows * Ncols) {
        int k = i / Ncols, n = i % Ncols;
        WT[(size_t)n * Krows + k] = f2h(W[i]);
    }
}

// ---------------------------------------------------------------------------
// Kernel 2: fused QKV GEMM.  Y = xb(4096x1024) @ W(1024x2048), z picks weight.
// Block tile 128x128, 8 waves as 2x4, wave tile 64x32 (4x2 WMMA tiles).
// z==2 (V) stores transposed per head into Vt.
__global__ __launch_bounds__(256) void csa_qkv_gemm_kernel(
    const unsigned short* __restrict__ xb,
    const unsigned short* __restrict__ WqT,
    const unsigned short* __restrict__ WkT,
    const unsigned short* __restrict__ WvT,
    unsigned short* __restrict__ Qo,
    unsigned short* __restrict__ Ko,
    unsigned short* __restrict__ Vt) {
    const int lane = threadIdx.x & 31, l16 = lane & 15, lhi = lane >> 4;
    const int w = threadIdx.x >> 5;
    const int Mbase = blockIdx.y * 128 + (w >> 2) * 64;
    const int Nbase = blockIdx.x * 128 + (w & 3) * 32;
    const unsigned short* BT = (blockIdx.z == 0) ? WqT
                             : (blockIdx.z == 1) ? WkT : WvT;
    v8f acc[4][2] = {};
    for (int k = 0; k < CIN; k += 32) {
        __builtin_prefetch(xb + (size_t)(Mbase + l16) * CIN + k + 64, 0, 3);
        Frag a[4], b[2];
#pragma unroll
        for (int mt = 0; mt < 4; ++mt) {
            const unsigned short* p =
                xb + (size_t)(Mbase + mt * 16 + l16) * CIN + k + lhi * 8;
            a[mt].q[0] = ldg128(p);
            a[mt].q[1] = ldg128(p + 16);
        }
#pragma unroll
        for (int nt = 0; nt < 2; ++nt) {
            const unsigned short* p =
                BT + (size_t)(Nbase + nt * 16 + l16) * CIN + k + lhi * 8;
            b[nt].q[0] = ldg128(p);
            b[nt].q[1] = ldg128(p + 16);
        }
#pragma unroll
        for (int mt = 0; mt < 4; ++mt)
#pragma unroll
            for (int nt = 0; nt < 2; ++nt)
                acc[mt][nt] = wmma16(a[mt], b[nt], acc[mt][nt]);
    }
    if (blockIdx.z < 2) {
        unsigned short* O = (blockIdx.z == 0) ? Qo : Ko;
#pragma unroll
        for (int mt = 0; mt < 4; ++mt)
#pragma unroll
            for (int nt = 0; nt < 2; ++nt)
#pragma unroll
                for (int i = 0; i < 8; ++i) {
                    int row = Mbase + mt * 16 + i + lhi * 8;
                    int col = Nbase + nt * 16 + l16;
                    O[(size_t)row * C2 + col] = f2h(acc[mt][nt][i]);
                }
    } else {
#pragma unroll
        for (int mt = 0; mt < 4; ++mt)
#pragma unroll
            for (int nt = 0; nt < 2; ++nt)
#pragma unroll
                for (int i = 0; i < 8; ++i) {
                    int row = Mbase + mt * 16 + i + lhi * 8;   // token
                    int col = Nbase + nt * 16 + l16;           // channel
                    int bb = row >> 11, nn = row & (SEQ - 1);
                    int h = col >> 7, d = col & 127;
                    Vt[(((size_t)(bb * HEADS + h) * 128 + d) << 11) + nn] =
                        f2h(acc[mt][nt][i]);
                }
    }
}

// ---------------------------------------------------------------------------
// Kernel 3: causal flash attention with async-LDS double buffering.
// grid = (N/128 query blocks, B*H); 8 waves, wave w owns rows [q0b+16w, +16).
// Per 32-key step: 256 threads issue 4x global_load_async_to_lds_b128 each
// (K tile 32x128 -> Kbuf[key][d], V tile 128x32 -> Vbuf[d][key]); compute on
// tile i overlaps the async DMA of tile i+1.  Online softmax in f32.
__global__ __launch_bounds__(256) void csa_attn_kernel(
    const unsigned short* __restrict__ Qg,
    const unsigned short* __restrict__ Kg,
    const unsigned short* __restrict__ Vt,
    unsigned short* __restrict__ A2) {
    __shared__ __align__(16) unsigned short Kbuf[2][32 * KPAD];
    __shared__ __align__(16) unsigned short Vbuf[2][128 * VPAD];
    __shared__ __align__(16) unsigned short Plds[8 * 16 * 32];

    const int tid = threadIdx.x;
    const int lane = tid & 31, l16 = lane & 15, lhi = lane >> 4;
    const int w = tid >> 5;
    const int bb = blockIdx.y >> 4, h = blockIdx.y & 15;
    const int q0b = blockIdx.x * 128;
    const int q0 = q0b + w * 16;

    const unsigned short* Qb = Qg + (size_t)bb * SEQ * C2 + h * 128;
    const unsigned short* Kb = Kg + (size_t)bb * SEQ * C2 + h * 128;
    const unsigned short* Vb = Vt + (size_t)(bb * HEADS + h) * 128 * SEQ;
    unsigned short*       Ob = A2 + (size_t)bb * SEQ * C2 + h * 128;
    unsigned short*       Pw = Plds + w * 512;

    // per-thread async chunk coordinates (16B chunks)
    const int kr = tid >> 4, kc = (tid & 15) * 8;   // K: rows kr, kr+16
    const int vd = tid >> 2, vc = (tid & 3) * 8;    // V: rows vd, vd+64

    Frag qf[4];
#pragma unroll
    for (int j = 0; j < 4; ++j) {
        const unsigned short* p = Qb + (size_t)(q0 + l16) * C2 + j * 32 + lhi * 8;
        qf[j].q[0] = ldg128(p);
        qf[j].q[1] = ldg128(p + 16);
    }

    v8f o[8] = {};
    float mrow[8], lrow[8];
#pragma unroll
    for (int i = 0; i < 8; ++i) { mrow[i] = -1e30f; lrow[i] = 0.0f; }

    const int niter = (q0b + 128) / 32;   // uniform across the block

    // prologue: async-issue tile 0 into buffer 0
    {
        unsigned kof = (unsigned)(size_t)&Kbuf[0][0];
        unsigned vof = (unsigned)(size_t)&Vbuf[0][0];
        async_ld128(kof + (unsigned)(kr * KPAD + kc) * 2,
                    Kb + (size_t)kr * C2 + kc);
        async_ld128(kof + (unsigned)((kr + 16) * KPAD + kc) * 2,
                    Kb + (size_t)(kr + 16) * C2 + kc);
        async_ld128(vof + (unsigned)(vd * VPAD + vc) * 2,
                    Vb + (size_t)vd * SEQ + vc);
        async_ld128(vof + (unsigned)((vd + 64) * VPAD + vc) * 2,
                    Vb + (size_t)(vd + 64) * SEQ + vc);
    }

    for (int it = 0; it < niter; ++it) {
        const int kv = it * 32;
        __syncthreads();                  // release buffer (it+1)&1
        if (it + 1 < niter) {
            const int kvn = kv + 32;
            unsigned kof = (unsigned)(size_t)&Kbuf[(it + 1) & 1][0];
            unsigned vof = (unsigned)(size_t)&Vbuf[(it + 1) & 1][0];
            async_ld128(kof + (unsigned)(kr * KPAD + kc) * 2,
                        Kb + (size_t)(kvn + kr) * C2 + kc);
            async_ld128(kof + (unsigned)((kr + 16) * KPAD + kc) * 2,
                        Kb + (size_t)(kvn + kr + 16) * C2 + kc);
            async_ld128(vof + (unsigned)(vd * VPAD + vc) * 2,
                        Vb + (size_t)vd * SEQ + kvn + vc);
            async_ld128(vof + (unsigned)((vd + 64) * VPAD + vc) * 2,
                        Vb + (size_t)(vd + 64) * SEQ + kvn + vc);
            // async loads complete in order: cnt<=4 => tile `it` has landed
            asm volatile("s_wait_asynccnt 0x4" ::: "memory");
        } else {
            asm volatile("s_wait_asynccnt 0x0" ::: "memory");
        }
        __syncthreads();                  // tile `it` visible to all waves

        const unsigned short* kt = Kbuf[it & 1];
        const unsigned short* vt = Vbuf[it & 1];

        // S = Q * K^T  (two 16x16 key tiles)
        v8f s0 = {}, s1 = {};
#pragma unroll
        for (int j = 0; j < 4; ++j) {     // d chunks of 32
            Frag k0, k1;
            const unsigned short* p0 = kt + l16 * KPAD + j * 32 + lhi * 8;
            k0.q[0] = *(const v4u*)p0;
            k0.q[1] = *(const v4u*)(p0 + 16);
            const unsigned short* p1 = p0 + 16 * KPAD;
            k1.q[0] = *(const v4u*)p1;
            k1.q[1] = *(const v4u*)(p1 + 16);
            s0 = wmma16(qf[j], k0, s0);
            s1 = wmma16(qf[j], k1, s1);
        }

        // online softmax update (per-VGPR-row state, 16-lane reductions)
#pragma unroll
        for (int i = 0; i < 8; ++i) {
            int row = q0 + i + lhi * 8;
            int c0 = kv + l16, c1 = kv + 16 + l16;
            float v0 = s0[i] * 0.125f;    // scale 1/sqrt(64)
            float v1 = s1[i] * 0.125f;
            if (c0 > row) v0 = -1e30f;
            if (c1 > row) v1 = -1e30f;
            float mx = fmaxf(v0, v1);
            for (int d = 1; d < 16; d <<= 1)
                mx = fmaxf(mx, __shfl_xor(mx, d, 32));
            float mnew  = fmaxf(mrow[i], mx);
            float scale = __expf(mrow[i] - mnew);
            float p0 = __expf(v0 - mnew), p1 = __expf(v1 - mnew);
            float rs = p0 + p1;
            for (int d = 1; d < 16; d <<= 1)
                rs += __shfl_xor(rs, d, 32);
            lrow[i] = lrow[i] * scale + rs;
            mrow[i] = mnew;
#pragma unroll
            for (int t = 0; t < 8; ++t) o[t][i] *= scale;
            int m = i + lhi * 8;          // C-layout row within tile
            Pw[m * 32 + l16]      = f2h(p0);
            Pw[m * 32 + 16 + l16] = f2h(p1);
        }
        // re-read P in A-fragment layout (same-wave LDS is in-order)
        Frag pf;
        pf.q[0] = *(const v4u*)(Pw + l16 * 32 + lhi * 8);
        pf.q[1] = *(const v4u*)(Pw + l16 * 32 + lhi * 8 + 16);
#pragma unroll
        for (int t = 0; t < 8; ++t) {     // O += P * V  (8 d-tiles)
            Frag vf;
            const unsigned short* p = vt + (t * 16 + l16) * VPAD + lhi * 8;
            vf.q[0] = *(const v4u*)p;
            vf.q[1] = *(const v4u*)(p + 16);
            o[t] = wmma16(pf, vf, o[t]);
        }
    }

#pragma unroll
    for (int i = 0; i < 8; ++i) {
        float inv = 1.0f / lrow[i];
        int row = q0 + i + lhi * 8;
#pragma unroll
        for (int t = 0; t < 8; ++t)
            Ob[(size_t)row * C2 + t * 16 + l16] = f2h(o[t][i] * inv);
    }
}

// ---------------------------------------------------------------------------
// Kernel 4: output projection.  out = A2(4096x2048) @ Wo(2048x1024), f32 out.
__global__ __launch_bounds__(256) void csa_oproj_gemm_kernel(
    const unsigned short* __restrict__ A2,
    const unsigned short* __restrict__ WoT,
    float* __restrict__ out) {
    const int lane = threadIdx.x & 31, l16 = lane & 15, lhi = lane >> 4;
    const int w = threadIdx.x >> 5;
    const int Mbase = blockIdx.y * 128 + (w >> 2) * 64;
    const int Nbase = blockIdx.x * 128 + (w & 3) * 32;
    v8f acc[4][2] = {};
    for (int k = 0; k < C2; k += 32) {
        __builtin_prefetch(A2 + (size_t)(Mbase + l16) * C2 + k + 64, 0, 3);
        Frag a[4], b[2];
#pragma unroll
        for (int mt = 0; mt < 4; ++mt) {
            const unsigned short* p =
                A2 + (size_t)(Mbase + mt * 16 + l16) * C2 + k + lhi * 8;
            a[mt].q[0] = ldg128(p);
            a[mt].q[1] = ldg128(p + 16);
        }
#pragma unroll
        for (int nt = 0; nt < 2; ++nt) {
            const unsigned short* p =
                WoT + (size_t)(Nbase + nt * 16 + l16) * C2 + k + lhi * 8;
            b[nt].q[0] = ldg128(p);
            b[nt].q[1] = ldg128(p + 16);
        }
#pragma unroll
        for (int mt = 0; mt < 4; ++mt)
#pragma unroll
            for (int nt = 0; nt < 2; ++nt)
                acc[mt][nt] = wmma16(a[mt], b[nt], acc[mt][nt]);
    }
#pragma unroll
    for (int mt = 0; mt < 4; ++mt)
#pragma unroll
        for (int nt = 0; nt < 2; ++nt)
#pragma unroll
            for (int i = 0; i < 8; ++i) {
                int row = Mbase + mt * 16 + i + lhi * 8;
                int col = Nbase + nt * 16 + l16;
                out[(size_t)row * CIN + col] = acc[mt][nt][i];
            }
}

// ---------------------------------------------------------------------------
extern "C" void kernel_launch(void* const* d_in, const int* in_sizes, int n_in,
                              void* d_out, int out_size, void* d_ws,
                              size_t ws_size, hipStream_t stream) {
    (void)in_sizes; (void)n_in; (void)out_size; (void)ws_size;
    const float* x  = (const float*)d_in[0];
    const float* Wq = (const float*)d_in[1];
    const float* Wk = (const float*)d_in[2];
    const float* Wv = (const float*)d_in[3];
    const float* Wo = (const float*)d_in[4];

    unsigned short* xb  = (unsigned short*)d_ws;            // 4M elems
    unsigned short* WqT = xb  + (size_t)NTOK * CIN;         // 2M
    unsigned short* WkT = WqT + (size_t)C2 * CIN;           // 2M
    unsigned short* WvT = WkT + (size_t)C2 * CIN;           // 2M
    unsigned short* WoT = WvT + (size_t)C2 * CIN;           // 2M
    unsigned short* Qb  = WoT + (size_t)CIN * C2;           // 8M
    unsigned short* Kb  = Qb  + (size_t)NTOK * C2;          // 8M
    unsigned short* Vt  = Kb  + (size_t)NTOK * C2;          // 8M
    unsigned short* A2  = Vt  + (size_t)2 * HEADS * 128 * SEQ; // 8M

    csa_cvt_kernel<<<(NTOK * CIN) / 256, 256, 0, stream>>>(x, xb, NTOK * CIN);
    csa_cvt_tr_kernel<<<(CIN * C2) / 256, 256, 0, stream>>>(Wq, WqT, CIN, C2);
    csa_cvt_tr_kernel<<<(CIN * C2) / 256, 256, 0, stream>>>(Wk, WkT, CIN, C2);
    csa_cvt_tr_kernel<<<(CIN * C2) / 256, 256, 0, stream>>>(Wv, WvT, CIN, C2);
    csa_cvt_tr_kernel<<<(C2 * CIN) / 256, 256, 0, stream>>>(Wo, WoT, C2, CIN);

    csa_qkv_gemm_kernel<<<dim3(C2 / 128, NTOK / 128, 3), 256, 0, stream>>>(
        xb, WqT, WkT, WvT, Qb, Kb, Vt);

    csa_attn_kernel<<<dim3(SEQ / 128, 2 * HEADS), 256, 0, stream>>>(
        Qb, Kb, Vt, A2);

    csa_oproj_gemm_kernel<<<dim3(CIN / 128, NTOK / 128), 256, 0, stream>>>(
        A2, WoT, (float*)d_out);
}
